// MultiHeadAttentionWeightSplits_72292889527137
// MI455X (gfx1250) — matrix operand order repaired and checked by reading, not compiled, not run on hardware
//
#include <hip/hip_runtime.h>

// MI455X / gfx1250 MHA forward.
// fp32 -> bf16 operands, v_wmma_f32_16x16x32_bf16 everywhere, fp32 accumulation.
// Workspace (u16 elements), 24M elems = 48 MB:
//   xb[4M] | Wq[1M] Wk[1M] Wv[1M] Wo[1M] | Q[4M] K[4M] V[4M] | ctx[4M]

#define DEV __device__ __forceinline__

typedef __attribute__((ext_vector_type(16))) __bf16 v16bf;
typedef __attribute__((ext_vector_type(8)))  __bf16 v8bf;
typedef __attribute__((ext_vector_type(8)))  float  v8f;

constexpr int D_EMB    = 1024;
constexpr int D_ATTN   = 1024;
constexpr int HEAD_DIM = 64;
constexpr int SEQ      = 2048;
constexpr int BS       = 2;
constexpr int NUM_HEADS= 16;
constexpr int MROWS    = BS * SEQ;   // 4096

DEV unsigned short f32_to_bf16(float f) {
  unsigned int u = __float_as_uint(f);
  u += 0x7FFFu + ((u >> 16) & 1u);   // round-to-nearest-even
  return (unsigned short)(u >> 16);
}

DEV v8f wmma_bf16(v16bf a, v16bf b, v8f c) {
  return __builtin_amdgcn_wmma_f32_16x16x32_bf16(false, a, false, b, (short)0, c,
                                                 false, false);
}

// ---- CDNA5 async global->LDS staging (ASYNCcnt path, CDNA5 ISA) --------
// vdst = LDS byte address (flat addr truncated per aperture rule addr[31:0]).
DEV void async_ld_b128(const unsigned short* g, unsigned short* l) {
  asm volatile("global_load_async_to_lds_b128 %0, %1, off"
               :: "v"((unsigned)(unsigned long long)l),
                  "v"((unsigned long long)g)
               : "memory");
}
DEV void wait_async0() {
#if __has_builtin(__builtin_amdgcn_s_wait_asynccnt)
  __builtin_amdgcn_s_wait_asynccnt(0);
#else
  asm volatile("s_wait_asynccnt 0x0" ::: "memory");
#endif
}

// A fragment: 16(M) x 32(K), row-major source, row stride lda (elements).
// Wave32: lane<16 -> M=lane, K {0..7,16..23}; lane>=16 -> K {8..15,24..31}.
DEV v16bf load_a_frag(const unsigned short* p, int lda) {
  const int lane = threadIdx.x & 31;
  const int m  = lane & 15;
  const int k0 = (lane & 16) ? 8 : 0;
  const unsigned short* r = p + (size_t)m * lda + k0;
  union { v16bf v; v8bf h[2]; } u;
  u.h[0] = *(const v8bf*)(r);
  u.h[1] = *(const v8bf*)(r + 16);
  return u.v;
}

// B fragment: 32(K) x 16(N), "N-major" source: element(k,n) = p[n*ldn + k].
// Wave32: lane<16 -> N=lane, K {0..7,16..23}; lane>=16 -> N=lane-16, K {8..15,24..31}.
DEV v16bf load_b_frag(const unsigned short* p, int ldn) {
  const int lane = threadIdx.x & 31;
  const int n  = lane & 15;
  const int k0 = (lane & 16) ? 8 : 0;
  const unsigned short* r = p + (size_t)n * ldn + k0;
  union { v16bf v; v8bf h[2]; } u;
  u.h[0] = *(const v8bf*)(r);
  u.h[1] = *(const v8bf*)(r + 16);
  return u.v;
}

// ---------------- fp32 -> bf16 conversion (x4 vectorized) -------------------
__global__ __launch_bounds__(256) void cvt_f32_bf16_v4(
    const float* __restrict__ in, unsigned short* __restrict__ out, int n4) {
  const int i = blockIdx.x * blockDim.x + threadIdx.x;
  if (i < n4) {
    const float4 f = ((const float4*)in)[i];
    union { unsigned short s[4]; uint2 v; } o;
    o.s[0] = f32_to_bf16(f.x);
    o.s[1] = f32_to_bf16(f.y);
    o.s[2] = f32_to_bf16(f.z);
    o.s[3] = f32_to_bf16(f.w);
    ((uint2*)out)[i] = o.v;
  }
}

// ---------------- C = A[M,K] @ W[N,K]^T  (torch Linear) ---------------------
// 256 threads = 8 waves (2x4), block tile 128x256, wave tile 64x64:
// 16 WMMA per 8 fragment loads per k-step (compute-bound friendly 2:1 ratio).
template <bool OUT_F32>
__global__ __launch_bounds__(256) void gemm_xwt(
    const unsigned short* __restrict__ A, const unsigned short* __restrict__ W,
    unsigned short* __restrict__ Cb, float* __restrict__ Cf,
    const float* __restrict__ bias, int M, int N, int K) {
  const int wave = threadIdx.x >> 5;
  const int m0 = blockIdx.x * 128 + (wave >> 2) * 64;
  const int n0 = blockIdx.y * 256 + (wave & 3) * 64;
  v8f acc[4][4] = {};
  for (int k = 0; k < K; k += 32) {
    __builtin_prefetch(A + (size_t)m0 * K + k + 256, 0, 1);
    __builtin_prefetch(W + (size_t)n0 * K + k + 256, 0, 1);
    v16bf af[4], bf[4];
#pragma unroll
    for (int i = 0; i < 4; ++i)
      af[i] = load_a_frag(A + (size_t)(m0 + 16 * i) * K + k, K);
#pragma unroll
    for (int i = 0; i < 4; ++i)
      bf[i] = load_b_frag(W + (size_t)(n0 + 16 * i) * K + k, K);
#pragma unroll
    for (int mi = 0; mi < 4; ++mi)
#pragma unroll
      for (int ni = 0; ni < 4; ++ni)
        acc[mi][ni] = wmma_bf16(af[mi], bf[ni], acc[mi][ni]);
  }
  const int lane = threadIdx.x & 31;
  const int nl   = lane & 15;
  const int moff = (lane & 16) ? 8 : 0;
#pragma unroll
  for (int mi = 0; mi < 4; ++mi)
#pragma unroll
    for (int ni = 0; ni < 4; ++ni) {
      const int col = n0 + ni * 16 + nl;
      const int rb  = m0 + mi * 16 + moff;
      const float bv = OUT_F32 ? bias[col] : 0.0f;
#pragma unroll
      for (int i = 0; i < 8; ++i) {
        if (OUT_F32) Cf[(size_t)(rb + i) * N + col] = acc[mi][ni][i] + bv;
        else         Cb[(size_t)(rb + i) * N + col] = f32_to_bf16(acc[mi][ni][i]);
      }
    }
}

// ---------------- causal flash attention ------------------------------------
// Grid: (BS*NUM_HEADS, SEQ/64). Block: 128 threads = 4 waves x 16 query rows.
__global__ __launch_bounds__(128) void attn_fwd(
    const unsigned short* __restrict__ Q, const unsigned short* __restrict__ Kt,
    const unsigned short* __restrict__ V, unsigned short* __restrict__ Ctx) {
  constexpr int KSTRIDE = 72;  // 64 + 8 pad (16B-aligned rows, de-banked)
  constexpr int VSTRIDE = 40;  // 32 + 8 pad
  __shared__ unsigned short Ks[32 * KSTRIDE];     // K tile, row-major [key][d]
  __shared__ unsigned short Vt[64 * VSTRIDE];     // V tile transposed [d][key]
  __shared__ unsigned short Ps[4][16 * 32];       // per-wave P scratch

  const int bh   = blockIdx.x;
  const int b    = bh >> 4, h = bh & 15;
  const int qblk = blockIdx.y;
  const int wave = threadIdx.x >> 5;
  const int lane = threadIdx.x & 31;
  const int qbase = qblk * 64 + wave * 16;
  const size_t base = (size_t)b * SEQ * D_ATTN + (size_t)h * HEAD_DIM;

  const v16bf qa0 = load_a_frag(Q + base + (size_t)qbase * D_ATTN, D_ATTN);
  const v16bf qa1 = load_a_frag(Q + base + (size_t)qbase * D_ATTN + 32, D_ATTN);

  v8f acc0 = {}, acc1 = {}, acc2 = {}, acc3 = {};
  float mrow[8], lrow[8];
#pragma unroll
  for (int i = 0; i < 8; ++i) { mrow[i] = -1e30f; lrow[i] = 0.0f; }

  const int nl   = lane & 15;
  const int moff = (lane & 16) ? 8 : 0;
  const float scale = 0.125f;           // 1/sqrt(64)
  const int jend = qblk * 2 + 2;        // 32-key steps covering causal range

  for (int j = 0; j < jend; ++j) {
    __syncthreads();                    // previous tile fully consumed
    {   // cooperative staging: 128 threads, 16 elems each for K and V
      const int t  = threadIdx.x;
      const int r  = t >> 2;            // key row 0..31
      const int cl = (t & 3) * 16;      // d col {0,16,32,48}
      // K tile: async DMA global -> LDS (ASYNCcnt), two b128 per thread
      const unsigned short* ksrc = Kt + base + (size_t)(j * 32 + r) * D_ATTN + cl;
      async_ld_b128(ksrc,     &Ks[r * KSTRIDE + cl]);
      async_ld_b128(ksrc + 8, &Ks[r * KSTRIDE + cl + 8]);
      // V tile: synchronous load + transposed scatter into Vt[d][key]
      const unsigned short* vsrc = V + base + (size_t)(j * 32 + r) * D_ATTN + cl;
      union { v8bf v; unsigned short s[8]; } v0, v1;
      v0.v = *(const v8bf*)(vsrc);
      v1.v = *(const v8bf*)(vsrc + 8);
#pragma unroll
      for (int i = 0; i < 8; ++i) {
        Vt[(cl + i)     * VSTRIDE + r] = v0.s[i];
        Vt[(cl + 8 + i) * VSTRIDE + r] = v1.s[i];
      }
    }
    wait_async0();                      // K tile landed in LDS
    __syncthreads();

    // S[16q x 32k] = Q(16x64) @ K^T : two N-tiles, K-dim 64 in two steps
    const v8f z = {};
    v8f s0 = wmma_bf16(qa0, load_b_frag(&Ks[0], KSTRIDE), z);
    s0     = wmma_bf16(qa1, load_b_frag(&Ks[32], KSTRIDE), s0);
    v8f s1 = wmma_bf16(qa0, load_b_frag(&Ks[16 * KSTRIDE], KSTRIDE), z);
    s1     = wmma_bf16(qa1, load_b_frag(&Ks[16 * KSTRIDE + 32], KSTRIDE), s1);

    // online softmax (rows in VGPR index; cols across 16-lane half-groups)
    const int kcol0 = j * 32 + nl;
    const int kcol1 = kcol0 + 16;
    unsigned short* pw = Ps[wave];
#pragma unroll
    for (int i = 0; i < 8; ++i) {
      const int qrow = qbase + moff + i;
      float x0 = (kcol0 <= qrow) ? s0[i] * scale : -1e30f;
      float x1 = (kcol1 <= qrow) ? s1[i] * scale : -1e30f;
      float mx = fmaxf(x0, x1);
#pragma unroll
      for (int msk = 8; msk >= 1; msk >>= 1) mx = fmaxf(mx, __shfl_xor(mx, msk, 32));
      const float mnew  = fmaxf(mrow[i], mx);
      const float alpha = __expf(mrow[i] - mnew);
      const float p0 = __expf(x0 - mnew);
      const float p1 = __expf(x1 - mnew);
      float ps = p0 + p1;
#pragma unroll
      for (int msk = 8; msk >= 1; msk >>= 1) ps += __shfl_xor(ps, msk, 32);
      lrow[i] = lrow[i] * alpha + ps;
      mrow[i] = mnew;
      acc0[i] *= alpha; acc1[i] *= alpha; acc2[i] *= alpha; acc3[i] *= alpha;
      pw[(moff + i) * 32 + nl]      = f32_to_bf16(p0);   // C-layout -> LDS
      pw[(moff + i) * 32 + nl + 16] = f32_to_bf16(p1);
    }

    // ctx += P(16x32) @ V(32x64): reload P in A-layout, V from transposed LDS
    const v16bf pa = load_a_frag(pw, 32);
    acc0 = wmma_bf16(pa, load_b_frag(&Vt[0],            VSTRIDE), acc0);
    acc1 = wmma_bf16(pa, load_b_frag(&Vt[16 * VSTRIDE], VSTRIDE), acc1);
    acc2 = wmma_bf16(pa, load_b_frag(&Vt[32 * VSTRIDE], VSTRIDE), acc2);
    acc3 = wmma_bf16(pa, load_b_frag(&Vt[48 * VSTRIDE], VSTRIDE), acc3);
  }

  unsigned short* cbase = Ctx + base;
#pragma unroll
  for (int i = 0; i < 8; ++i) {
    const float inv = 1.0f / lrow[i];
    unsigned short* dst = cbase + (size_t)(qbase + moff + i) * D_ATTN + nl;
    dst[0]  = f32_to_bf16(acc0[i] * inv);
    dst[16] = f32_to_bf16(acc1[i] * inv);
    dst[32] = f32_to_bf16(acc2[i] * inv);
    dst[48] = f32_to_bf16(acc3[i] * inv);
  }
}

// ---------------------------------------------------------------------------
extern "C" void kernel_launch(void* const* d_in, const int* in_sizes, int n_in,
                              void* d_out, int out_size, void* d_ws, size_t ws_size,
                              hipStream_t stream) {
  const float* x  = (const float*)d_in[0];
  const float* Wq = (const float*)d_in[1];
  const float* Wk = (const float*)d_in[2];
  const float* Wv = (const float*)d_in[3];
  const float* Wo = (const float*)d_in[4];
  const float* bo = (const float*)d_in[5];

  unsigned short* ws = (unsigned short*)d_ws;   // needs 48 MB
  const size_t NX = (size_t)MROWS * D_EMB;      // 4M elems
  const size_t NW = (size_t)D_ATTN * D_EMB;     // 1M elems
  unsigned short* xb  = ws;
  unsigned short* wqb = xb  + NX;
  unsigned short* wkb = wqb + NW;
  unsigned short* wvb = wkb + NW;
  unsigned short* wob = wvb + NW;
  unsigned short* qb  = wob + NW;
  unsigned short* kb  = qb  + NX;
  unsigned short* vb  = kb  + NX;
  unsigned short* ctx = vb  + NX;

  {
    int n4 = (int)(NX / 4);
    cvt_f32_bf16_v4<<<(n4 + 255) / 256, 256, 0, stream>>>(x, xb, n4);
    n4 = (int)(NW / 4);
    cvt_f32_bf16_v4<<<(n4 + 255) / 256, 256, 0, stream>>>(Wq, wqb, n4);
    cvt_f32_bf16_v4<<<(n4 + 255) / 256, 256, 0, stream>>>(Wk, wkb, n4);
    cvt_f32_bf16_v4<<<(n4 + 255) / 256, 256, 0, stream>>>(Wv, wvb, n4);
    cvt_f32_bf16_v4<<<(n4 + 255) / 256, 256, 0, stream>>>(Wo, wob, n4);
  }

  const dim3 g(MROWS / 128, D_ATTN / 256);   // (32, 4)
  gemm_xwt<false><<<g, 256, 0, stream>>>(xb, wqb, qb, nullptr, nullptr,
                                         MROWS, D_ATTN, D_EMB);
  gemm_xwt<false><<<g, 256, 0, stream>>>(xb, wkb, kb, nullptr, nullptr,
                                         MROWS, D_ATTN, D_EMB);
  gemm_xwt<false><<<g, 256, 0, stream>>>(xb, wvb, vb, nullptr, nullptr,
                                         MROWS, D_ATTN, D_EMB);

  attn_fwd<<<dim3(BS * NUM_HEADS, SEQ / 64), 128, 0, stream>>>(qb, kb, vb, ctx);

  gemm_xwt<true><<<g, 256, 0, stream>>>(ctx, wob, nullptr, (float*)d_out, bo,
                                        MROWS, D_ATTN, D_ATTN);
}
